// EncoderBlock_69063074120146
// MI455X (gfx1250) — matrix operand order repaired and checked by reading, not compiled, not run on hardware
//
#include <hip/hip_runtime.h>
#include <hip/hip_bf16.h>

// Problem constants (from reference)
#define B_   2
#define S_   2048
#define D_   1024
#define H_   16
#define FF_  4096
#define HD_  64
#define MROWS (B_*S_)   // 4096 token rows

typedef __bf16 bf16;
typedef __bf16        v16bf __attribute__((ext_vector_type(16)));
typedef float         v8f   __attribute__((ext_vector_type(8)));
typedef unsigned int  u32x4 __attribute__((ext_vector_type(4)));

union FragBF { v16bf v; u32x4 q[2]; };

static __device__ __forceinline__ v8f vzero() {
  v8f z;
#pragma unroll
  for (int i = 0; i < 8; ++i) z[i] = 0.0f;
  return z;
}

static __device__ __forceinline__ v8f wmma_bf16(v16bf a, v16bf b, v8f c) {
  // D = A(16x32 bf16) * B(32x16 bf16) + C(16x16 f32)
  return __builtin_amdgcn_wmma_f32_16x16x32_bf16(false, a, false, b,
                                                 (short)0, c, false, false);
}

// CDNA5 async copy: global -> LDS, 16 bytes per lane, tracked by ASYNCcnt.
// LDS destination VGPR holds the wave-relative LDS byte offset, which equals
// the low 32 bits of the generic pointer to a __shared__ object (ISA 10.2:
// LDS aperture maps addr[31:0] to the LDS offset).
static __device__ __forceinline__ void async_copy_b128(const bf16* lds_dst,
                                                       const bf16* gsrc) {
  unsigned lds_off = (unsigned)(size_t)lds_dst;
  unsigned long long gaddr = (unsigned long long)(size_t)gsrc;
  asm volatile("global_load_async_to_lds_b128 %0, %1, off"
               :: "v"(lds_off), "v"(gaddr) : "memory");
}
static __device__ __forceinline__ void wait_async0() {
  asm volatile("s_wait_asynccnt 0" ::: "memory");
}

// A fragment (16x32) from row-major [rows][ldk] bf16 storage.
// ISA layout: lanes 0-15 rows 0-15, VGPR0-3 = K (half*8 .. +7), VGPR4-7 = K+16.
static __device__ __forceinline__ FragBF load_a(const bf16* base, int ldk, int lane) {
  const int row = lane & 15, khalf = lane >> 4;
  FragBF f;
  const bf16* p = base + row * ldk + khalf * 8;
  f.q[0] = *(const u32x4*)(p);
  f.q[1] = *(const u32x4*)(p + 16);
  return f;
}

// B fragment (32x16, KxN) from TRANSPOSED [N][ldk] bf16 storage.
// ISA layout: lane n holds column n; VGPR0-3 = K (khalf*8 .. +7), VGPR4-7 = K+16.
static __device__ __forceinline__ FragBF load_b(const bf16* base, int ldk, int lane) {
  const int n = lane & 15, khalf = lane >> 4;
  FragBF f;
  const bf16* p = base + n * ldk + khalf * 8;
  f.q[0] = *(const u32x4*)(p);
  f.q[1] = *(const u32x4*)(p + 16);
  return f;
}

// ---------------------------------------------------------------------------
// Generic bf16 GEMM, C[M,N] = A[M,K] @ Bt^T + bias[n]  (+epilogue)
// Bt is PRE-TRANSPOSED [N][K] so both operands stage with contiguous
// async-to-LDS b128 copies (no scalar transpose scatter).
// Block tile 128x128, 8 waves (2 along M x 4 along N), wave tile 64x32.
// ---------------------------------------------------------------------------
constexpr int FLAG_BF16 = 1;   // output bf16 (else f32)
constexpr int FLAG_RELU = 2;
constexpr int FLAG_RES  = 4;   // add f32 residual[M,N]
constexpr int FLAG_VT   = 8;   // scatter bf16 output to [B][H][HD][S] (V path)

template <int FLAGS>
__global__ __launch_bounds__(256)
void gemm_bf16_kernel(const bf16* __restrict__ A, const bf16* __restrict__ Bt,
                      const float* __restrict__ bias, const float* __restrict__ resid,
                      void* __restrict__ Cout, int M, int N, int K)
{
  __shared__ bf16 ldsA[128 * 64];    // [m][k]
  __shared__ bf16 ldsBt[128 * 64];   // [n][k]

  const int tid  = threadIdx.x;
  const int lane = tid & 31;
  const int w    = tid >> 5;    // 8 waves
  const int wm   = w & 1;       // 2 waves along M
  const int wn   = w >> 1;      // 4 waves along N
  const int m0   = blockIdx.y * 128;
  const int n0   = blockIdx.x * 128;

  v8f acc[4][2];
#pragma unroll
  for (int mt = 0; mt < 4; ++mt)
#pragma unroll
    for (int nt = 0; nt < 2; ++nt) acc[mt][nt] = vzero();

  // staging coordinates: thread covers one 32-element half-row of each tile
  const int arow  = tid >> 1;          // 0..127
  const int apart = (tid & 1) * 32;    // 0 / 32

  for (int k0 = 0; k0 < K; k0 += 64) {
    { // stage A tile 128x64 (async, 4x b128)
      const bf16* g = A + (size_t)(m0 + arow) * K + k0 + apart;
      const bf16* l = ldsA + arow * 64 + apart;
#pragma unroll
      for (int c = 0; c < 4; ++c) async_copy_b128(l + c * 8, g + c * 8);
    }
    { // stage Bt tile 128x64 (async, 4x b128) -- already [n][k]
      const bf16* g = Bt + (size_t)(n0 + arow) * K + k0 + apart;
      const bf16* l = ldsBt + arow * 64 + apart;
#pragma unroll
      for (int c = 0; c < 4; ++c) async_copy_b128(l + c * 8, g + c * 8);
    }
    if (k0 + 64 < K) {  // hint next tiles into cache (global_prefetch_b8)
      __builtin_prefetch(A + (size_t)(m0 + arow) * K + k0 + 64 + apart, 0, 1);
      __builtin_prefetch(Bt + (size_t)(n0 + arow) * K + k0 + 64 + apart, 0, 1);
    }
    wait_async0();
    __syncthreads();

#pragma unroll
    for (int kk = 0; kk < 2; ++kk) {       // two 32-deep WMMA steps
      FragBF a[4];
#pragma unroll
      for (int mt = 0; mt < 4; ++mt)
        a[mt] = load_a(ldsA + (wm * 64 + mt * 16) * 64 + kk * 32, 64, lane);
#pragma unroll
      for (int nt = 0; nt < 2; ++nt) {
        FragBF b = load_b(ldsBt + (wn * 32 + nt * 16) * 64 + kk * 32, 64, lane);
#pragma unroll
        for (int mt = 0; mt < 4; ++mt)
          acc[mt][nt] = wmma_bf16(a[mt].v, b.v, acc[mt][nt]);
      }
    }
    __syncthreads();
  }

  // Epilogue: C layout = N across lanes, M = r + 8*(lane>=16)
  const int lh = lane >> 4, nl = lane & 15;
#pragma unroll
  for (int mt = 0; mt < 4; ++mt)
#pragma unroll
    for (int nt = 0; nt < 2; ++nt)
#pragma unroll
      for (int r = 0; r < 8; ++r) {
        const int gm = m0 + wm * 64 + mt * 16 + r + 8 * lh;
        const int gn = n0 + wn * 32 + nt * 16 + nl;
        float v = acc[mt][nt][r] + bias[gn];
        if (FLAGS & FLAG_RELU) v = fmaxf(v, 0.0f);
        if (FLAGS & FLAG_RES)  v += resid[(size_t)gm * N + gn];
        if (FLAGS & FLAG_VT) {
          // scatter to per-head transposed V: [B][H][HD][S]
          const int bb = gm >> 11, ss = gm & (S_ - 1);
          const int hh = gn >> 6,  dd = gn & (HD_ - 1);
          ((bf16*)Cout)[(((size_t)bb * H_ + hh) * HD_ + dd) * S_ + ss] = (bf16)v;
        } else if (FLAGS & FLAG_BF16) {
          ((bf16*)Cout)[(size_t)gm * N + gn] = (bf16)v;
        } else {
          ((float*)Cout)[(size_t)gm * N + gn] = v;
        }
      }
}

// ---------------------------------------------------------------------------
// Fused flash attention: per block one (b,h) and 128 query rows.
// 8 waves, wave w owns 16 query rows. Online softmax, O kept in registers.
// K tile arrives row-major [t][d] (= [N][K] for QK^T); V arrives from the
// pre-transposed [B][H][HD][S] buffer so both stage as contiguous async b128.
// ---------------------------------------------------------------------------
__global__ __launch_bounds__(256)
void attention_kernel(const bf16* __restrict__ Qb, const bf16* __restrict__ Kb,
                      const bf16* __restrict__ Vt, bf16* __restrict__ Cat)
{
  __shared__ bf16 ldsK[128 * 64];      // [t][d]
  __shared__ bf16 ldsVt[64 * 128];     // [d][t]
  __shared__ bf16 ldsP[8 * 16 * 128];  // per-wave P scratch (C->A layout round trip)

  const int tid = threadIdx.x, lane = tid & 31, w = tid >> 5;
  const int bh = blockIdx.y;
  const int b = bh >> 4, h = bh & 15;
  const int s0 = blockIdx.x * 128;

  const bf16* qbase  = Qb + (size_t)(b * S_) * D_ + (size_t)h * HD_;
  const bf16* kbase  = Kb + (size_t)(b * S_) * D_ + (size_t)h * HD_;
  const bf16* vtbase = Vt + ((size_t)(b * H_ + h) * HD_) * S_;

  // Q fragments for this wave's 16 rows (HD=64 -> two 32-deep fragments)
  FragBF qf[2];
  {
    const int row = lane & 15, khalf = lane >> 4;
    const bf16* qp = qbase + (size_t)(s0 + w * 16 + row) * D_;
#pragma unroll
    for (int kk = 0; kk < 2; ++kk) {
      const bf16* p = qp + kk * 32 + khalf * 8;
      qf[kk].q[0] = *(const u32x4*)(p);
      qf[kk].q[1] = *(const u32x4*)(p + 16);
    }
  }

  v8f Oacc[4];
#pragma unroll
  for (int nt = 0; nt < 4; ++nt) Oacc[nt] = vzero();
  float mrun[8], lrun[8];
#pragma unroll
  for (int r = 0; r < 8; ++r) { mrun[r] = -1e30f; lrun[r] = 0.0f; }
  const float scale = 0.125f;   // 1/sqrt(HD)

  const int krow  = tid >> 1;          // 0..127  (K staging)
  const int kpart = (tid & 1) * 32;
  const int vrow  = tid >> 2;          // 0..63   (V staging)
  const int vpart = (tid & 3) * 32;

  for (int t0 = 0; t0 < S_; t0 += 128) {
    { // stage K rows [t][d] (async, 4x b128)
      const bf16* g = kbase + (size_t)(t0 + krow) * D_ + kpart;
      const bf16* l = ldsK + krow * 64 + kpart;
#pragma unroll
      for (int c = 0; c < 4; ++c) async_copy_b128(l + c * 8, g + c * 8);
    }
    { // stage V^T rows [d][t] straight from [B][H][HD][S] (async, 4x b128)
      const bf16* g = vtbase + (size_t)vrow * S_ + t0 + vpart;
      const bf16* l = ldsVt + vrow * 128 + vpart;
#pragma unroll
      for (int c = 0; c < 4; ++c) async_copy_b128(l + c * 8, g + c * 8);
    }
    if (t0 + 128 < S_) {
      __builtin_prefetch(kbase + (size_t)(t0 + 128 + krow) * D_ + kpart, 0, 1);
      __builtin_prefetch(vtbase + (size_t)vrow * S_ + t0 + 128 + vpart, 0, 1);
    }
    wait_async0();
    __syncthreads();

    // scores S = Q @ K^T : 16 x 128, 8 n-tiles, 2 k-steps
    v8f sacc[8];
#pragma unroll
    for (int nt = 0; nt < 8; ++nt) sacc[nt] = vzero();
#pragma unroll
    for (int kk = 0; kk < 2; ++kk)
#pragma unroll
      for (int nt = 0; nt < 8; ++nt) {
        FragBF bkf = load_b(ldsK + (nt * 16) * 64 + kk * 32, 64, lane);
        sacc[nt] = wmma_bf16(qf[kk].v, bkf.v, sacc[nt]);
      }

    // online softmax update; write P (bf16) to per-wave LDS in [16][128]
    const int lh = lane >> 4, nl = lane & 15;
    bf16* pbase = ldsP + w * 16 * 128;
#pragma unroll
    for (int r = 0; r < 8; ++r) {
      float pm = -1e30f;
#pragma unroll
      for (int nt = 0; nt < 8; ++nt) pm = fmaxf(pm, sacc[nt][r] * scale);
#pragma unroll
      for (int off = 8; off; off >>= 1) pm = fmaxf(pm, __shfl_xor(pm, off, 32));
      const float mnew  = fmaxf(mrun[r], pm);
      const float alpha = __expf(mrun[r] - mnew);
      float ps = 0.0f;
#pragma unroll
      for (int nt = 0; nt < 8; ++nt) {
        const float p = __expf(sacc[nt][r] * scale - mnew);
        ps += p;
        pbase[(r + 8 * lh) * 128 + nt * 16 + nl] = (bf16)p;
      }
#pragma unroll
      for (int off = 8; off; off >>= 1) ps += __shfl_xor(ps, off, 32);
      lrun[r] = lrun[r] * alpha + ps;
      mrun[r] = mnew;
#pragma unroll
      for (int nt = 0; nt < 4; ++nt) Oacc[nt][r] *= alpha;
    }

    // O += P @ V  (P from per-wave LDS; same-wave LDS ordering is in-order)
#pragma unroll
    for (int kk = 0; kk < 4; ++kk) {
      FragBF ap = load_a(pbase + kk * 32, 128, lane);
#pragma unroll
      for (int nt = 0; nt < 4; ++nt) {
        FragBF bv = load_b(ldsVt + (nt * 16) * 128 + kk * 32, 128, lane);
        Oacc[nt] = wmma_bf16(ap.v, bv.v, Oacc[nt]);
      }
    }
    __syncthreads();
  }

  // normalize and write concatenated heads [B*S, H*HD] as bf16
  const int lh = lane >> 4, nl = lane & 15;
#pragma unroll
  for (int nt = 0; nt < 4; ++nt)
#pragma unroll
    for (int r = 0; r < 8; ++r) {
      const int grow = b * S_ + s0 + w * 16 + r + 8 * lh;
      const int gcol = h * HD_ + nt * 16 + nl;
      Cat[(size_t)grow * D_ + gcol] = (bf16)(Oacc[nt][r] / lrun[r]);
    }
}

// ---------------------------------------------------------------------------
// LayerNorm: one 256-thread block per row of D=1024. Unbiased var (ddof=1).
// ---------------------------------------------------------------------------
__global__ __launch_bounds__(256)
void layernorm_kernel(const float* __restrict__ in, const float* __restrict__ g,
                      const float* __restrict__ be, float* __restrict__ outf,
                      bf16* __restrict__ outb)
{
  __shared__ float red[16];
  __shared__ float stats[2];
  const int row = blockIdx.x, tid = threadIdx.x, lane = tid & 31, w = tid >> 5;
  const float* p = in + (size_t)row * D_;
  float s = 0.0f, sq = 0.0f, vals[4];
#pragma unroll
  for (int i = 0; i < 4; ++i) {
    const float v = p[tid + i * 256];
    vals[i] = v; s += v; sq += v * v;
  }
#pragma unroll
  for (int off = 16; off; off >>= 1) {
    s  += __shfl_xor(s,  off, 32);
    sq += __shfl_xor(sq, off, 32);
  }
  if (lane == 0) { red[w] = s; red[8 + w] = sq; }
  __syncthreads();
  if (tid == 0) {
    float ts = 0.0f, tq = 0.0f;
    for (int i = 0; i < 8; ++i) { ts += red[i]; tq += red[8 + i]; }
    const float mean = ts / (float)D_;
    const float var  = (tq - ts * ts / (float)D_) / (float)(D_ - 1);
    stats[0] = mean;
    stats[1] = rsqrtf(var + 1e-12f);
  }
  __syncthreads();
  const float mean = stats[0], rstd = stats[1];
#pragma unroll
  for (int i = 0; i < 4; ++i) {
    const int c = tid + i * 256;
    const float y = g[c] * (vals[i] - mean) * rstd + be[c];
    outf[(size_t)row * D_ + c] = y;
    if (outb) outb[(size_t)row * D_ + c] = (bf16)y;
  }
}

// ---------------------------------------------------------------------------
// One-time conversions (f32 -> bf16), weights pre-transposed to [N][K]
// ---------------------------------------------------------------------------
__global__ void cvt_f32_bf16(const float* __restrict__ in, bf16* __restrict__ out, int n) {
  const int i = blockIdx.x * 256 + threadIdx.x;
  if (i < n) out[i] = (bf16)in[i];
}

// generic weight [K, N] f32 -> [N, K] bf16
__global__ void cvt_w_t(const float* __restrict__ in, bf16* __restrict__ out,
                        int K, int N) {
  const int i = blockIdx.x * 256 + threadIdx.x;  // over N*K (out-linear)
  const int n = i / K, k = i % K;
  out[i] = (bf16)in[(size_t)k * N + n];
}

// head weights [H, D, HD] f32 -> [N=H*HD, K=D] bf16
__global__ void cvt_headw_t(const float* __restrict__ in, bf16* __restrict__ out) {
  const int i = blockIdx.x * 256 + threadIdx.x;  // over (H*HD)*D
  const int n = i >> 10, d = i & 1023;           // n = h*HD + kk
  const int h = n >> 6, kk = n & 63;
  out[i] = (bf16)in[((size_t)h * D_ + d) * HD_ + kk];
}

// ---------------------------------------------------------------------------
extern "C" void kernel_launch(void* const* d_in, const int* in_sizes, int n_in,
                              void* d_out, int out_size, void* d_ws, size_t ws_size,
                              hipStream_t stream)
{
  (void)in_sizes; (void)n_in; (void)out_size; (void)ws_size;
  const float* x   = (const float*)d_in[0];
  const float* Wq  = (const float*)d_in[1];
  const float* bq  = (const float*)d_in[2];
  const float* Wk  = (const float*)d_in[3];
  const float* bk  = (const float*)d_in[4];
  const float* Wv  = (const float*)d_in[5];
  const float* bv  = (const float*)d_in[6];
  const float* Wo  = (const float*)d_in[7];
  const float* bo  = (const float*)d_in[8];
  const float* g1  = (const float*)d_in[9];
  const float* b1  = (const float*)d_in[10];
  const float* W1  = (const float*)d_in[11];
  const float* bf1 = (const float*)d_in[12];
  const float* W2  = (const float*)d_in[13];
  const float* bf2 = (const float*)d_in[14];
  const float* g2  = (const float*)d_in[15];
  const float* b2  = (const float*)d_in[16];

  char* ws = (char*)d_ws;
  const size_t MB = 1u << 20;
  bf16*  WQB  = (bf16*)(ws +   0 * MB);   // 2 MB  [N=1024][K=1024]
  bf16*  WKB  = (bf16*)(ws +   2 * MB);   // 2 MB
  bf16*  WVB  = (bf16*)(ws +   4 * MB);   // 2 MB
  bf16*  WOB  = (bf16*)(ws +   6 * MB);   // 2 MB  [N=1024][K=1024]
  bf16*  W1B  = (bf16*)(ws +   8 * MB);   // 8 MB  [N=4096][K=1024]
  bf16*  W2B  = (bf16*)(ws +  16 * MB);   // 8 MB  [N=1024][K=4096]
  bf16*  XB   = (bf16*)(ws +  24 * MB);   // 8 MB  [M=4096][K=1024]
  bf16*  QB   = (bf16*)(ws +  32 * MB);   // 8 MB  [B*S][H*HD]
  bf16*  KB   = (bf16*)(ws +  40 * MB);   // 8 MB  [B*S][H*HD]
  bf16*  VTB  = (bf16*)(ws +  48 * MB);   // 8 MB  [B][H][HD][S]  (pre-transposed)
  bf16*  CAT  = (bf16*)(ws +  56 * MB);   // 8 MB
  float* MHAR = (float*)(ws +  64 * MB);  // 16 MB
  float* HF   = (float*)(ws +  80 * MB);  // 16 MB
  bf16*  HB   = (bf16*)(ws +  96 * MB);   // 8 MB
  bf16*  FF1  = (bf16*)(ws + 104 * MB);   // 32 MB
  float* FF2R = (float*)(ws + 136 * MB);  // 16 MB  (total 152 MB)

  const dim3 blk(256);
  cvt_f32_bf16<<<(MROWS * D_) / 256, blk, 0, stream>>>(x, XB, MROWS * D_);
  cvt_headw_t <<<(D_ * D_)   / 256, blk, 0, stream>>>(Wq, WQB);
  cvt_headw_t <<<(D_ * D_)   / 256, blk, 0, stream>>>(Wk, WKB);
  cvt_headw_t <<<(D_ * D_)   / 256, blk, 0, stream>>>(Wv, WVB);
  cvt_w_t     <<<(D_ * D_)   / 256, blk, 0, stream>>>(Wo, WOB, D_, D_);
  cvt_w_t     <<<(D_ * FF_)  / 256, blk, 0, stream>>>(W1, W1B, D_, FF_);
  cvt_w_t     <<<(FF_ * D_)  / 256, blk, 0, stream>>>(W2, W2B, FF_, D_);

  const dim3 gq(D_ / 128, MROWS / 128);    // (8, 32)
  gemm_bf16_kernel<FLAG_BF16><<<gq, blk, 0, stream>>>(XB, WQB, bq, nullptr, QB, MROWS, D_, D_);
  gemm_bf16_kernel<FLAG_BF16><<<gq, blk, 0, stream>>>(XB, WKB, bk, nullptr, KB, MROWS, D_, D_);
  gemm_bf16_kernel<FLAG_VT>  <<<gq, blk, 0, stream>>>(XB, WVB, bv, nullptr, VTB, MROWS, D_, D_);

  const dim3 ga(S_ / 128, B_ * H_);        // (16, 32)
  attention_kernel<<<ga, blk, 0, stream>>>(QB, KB, VTB, CAT);

  gemm_bf16_kernel<FLAG_RES><<<gq, blk, 0, stream>>>(CAT, WOB, bo, x, MHAR, MROWS, D_, D_);
  layernorm_kernel<<<MROWS, blk, 0, stream>>>(MHAR, g1, b1, HF, HB);

  const dim3 gf1(FF_ / 128, MROWS / 128);  // (32, 32)
  gemm_bf16_kernel<FLAG_BF16 | FLAG_RELU><<<gf1, blk, 0, stream>>>(HB, W1B, bf1, nullptr, FF1, MROWS, FF_, D_);
  gemm_bf16_kernel<FLAG_RES><<<gq, blk, 0, stream>>>(FF1, W2B, bf2, HF, FF2R, MROWS, D_, FF_);
  layernorm_kernel<<<MROWS, blk, 0, stream>>>(FF2R, g2, b2, (float*)d_out, nullptr);
}
